// NodeNetwork_83030307766411
// MI455X (gfx1250) — matrix operand order repaired and checked by reading, not compiled, not run on hardware
//
#include <hip/hip_runtime.h>
#include <hip/hip_bf16.h>
#include <math.h>

typedef __attribute__((ext_vector_type(16))) _Float16 v16h;
typedef __attribute__((ext_vector_type(8)))  _Float16 v8h;
typedef __attribute__((ext_vector_type(8)))  float    v8f;

#define NN 50000
#define EE 1600000
#define DD 32
#define HH 32
#define NTILES (NN / 16)      // 3125, exact
#define NFRAGS 12             // prebuilt B fragments

static __device__ __forceinline__ v8f wmma_f16(v16h a, v16h b, v8f c) {
  // D = A(16x32 f16) * B(32x16 f16) + C(16x16 f32)
  return __builtin_amdgcn_wmma_f32_16x16x32_f16(
      /*neg_a=*/false, a, /*neg_b=*/false, b,
      /*c_mod=*/(short)0, c, /*reuse_a=*/false, /*reuse_b=*/false);
}

// Branchless tanh via hardware v_exp_f32: tanh(x) = 1 - 2/(exp(2x)+1).
// Saturates exactly to +/-1 for large |x|; no EXEC manipulation.
static __device__ __forceinline__ float fast_tanh(float x) {
  const float e = __expf(2.0f * x);
  return 1.0f - 2.0f / (e + 1.0f);
}

// ---------------------------------------------------------------------------
// 1) zero the mi/mo accumulators (float4 stores)
// ---------------------------------------------------------------------------
__global__ void __launch_bounds__(256)
zero_kernel(float4* __restrict__ p, int n4) {
  int i = blockIdx.x * blockDim.x + threadIdx.x;
  if (i < n4) p[i] = make_float4(0.f, 0.f, 0.f, 0.f);
}

// ---------------------------------------------------------------------------
// 2) build the 12 B-matrix register fragments ONCE, in lane-major layout:
//    frags[((f*32)+lane)*16 + i] = b[i] for lane `lane` of fragment f.
//    B layout: lane = column; halfs 0-7 = K 8h..8h+7, halfs 8-15 = K 16+8h..16+8h+7
//    (h = lane>>4), matching the A fragment K order.
// ---------------------------------------------------------------------------
__global__ void __launch_bounds__(384)
build_frags_kernel(const float* __restrict__ W1, const float* __restrict__ W2,
                   const float* __restrict__ W3, const float* __restrict__ W4,
                   _Float16* __restrict__ frags) {
  const int f    = threadIdx.x >> 5;   // fragment 0..11
  const int lane = threadIdx.x & 31;
  if (f >= NFRAGS) return;

  const float* W; int k0, n0;
  switch (f) {
    case 0:  W = W1; k0 = 0;  n0 = 0;  break;
    case 1:  W = W1; k0 = 32; n0 = 0;  break;
    case 2:  W = W1; k0 = 64; n0 = 0;  break;
    case 3:  W = W1; k0 = 0;  n0 = 16; break;
    case 4:  W = W1; k0 = 32; n0 = 16; break;
    case 5:  W = W1; k0 = 64; n0 = 16; break;
    case 6:  W = W2; k0 = 0;  n0 = 0;  break;
    case 7:  W = W2; k0 = 0;  n0 = 16; break;
    case 8:  W = W3; k0 = 0;  n0 = 0;  break;
    case 9:  W = W3; k0 = 0;  n0 = 16; break;
    case 10: W = W4; k0 = 0;  n0 = 0;  break;
    default: W = W4; k0 = 0;  n0 = 16; break;
  }
  const int col = n0 + (lane & 15);
  const int kb  = k0 + (lane >> 4) * 8;
  _Float16* o = frags + ((size_t)(f * 32 + lane)) * 16;
#pragma unroll
  for (int i = 0; i < 16; ++i)
    o[i] = (_Float16)W[(size_t)(kb + i + (i >= 8 ? 8 : 0)) * HH + col];
}

// ---------------------------------------------------------------------------
// 3) edge gather-scale-scatter: one wave32 per edge, lane = feature channel.
//    Coalesced 128B row gathers; f32 global atomics into L2-resident mi/mo.
// ---------------------------------------------------------------------------
__global__ void __launch_bounds__(256)
edge_scatter_kernel(const float* __restrict__ x, const float* __restrict__ e,
                    const int* __restrict__ src, const int* __restrict__ dst,
                    float* __restrict__ mi, float* __restrict__ mo) {
  const int lane   = threadIdx.x & 31;
  const int wave   = blockIdx.x * (blockDim.x >> 5) + (threadIdx.x >> 5);
  const int nwaves = gridDim.x * (blockDim.x >> 5);
  for (int ed = wave; ed < EE; ed += nwaves) {
    const int   s  = src[ed];
    const int   d  = dst[ed];
    const float ev = e[ed];
    const float xs = x[(size_t)s * DD + lane];
    const float xd = x[(size_t)d * DD + lane];
    atomicAdd(&mi[(size_t)d * DD + lane], ev * xs);   // messages into dst
    atomicAdd(&mo[(size_t)s * DD + lane], ev * xd);   // messages into src
  }
}

// ---------------------------------------------------------------------------
// 4) 4-layer MLP with v_wmma_f32_16x16x32_f16. One wave per 16-node tile,
//    8 waves (256 threads) per block. B fragments are single 32B/lane loads.
// ---------------------------------------------------------------------------
__global__ void __launch_bounds__(256)
mlp_wmma_kernel(const float* __restrict__ mi, const float* __restrict__ mo,
                const float* __restrict__ x,
                const _Float16* __restrict__ frags,
                const float* __restrict__ b1, const float* __restrict__ b2,
                const float* __restrict__ b3, const float* __restrict__ b4,
                float* __restrict__ out) {
  __shared__ _Float16 hbuf[8][16 * HH];   // per-wave 16x32 activation tile

  const int lane    = threadIdx.x & 31;
  const int wid     = threadIdx.x >> 5;
  const int halfSel = lane >> 4;          // 0: lanes 0-15, 1: lanes 16-31
  const int mn      = lane & 15;          // A-row / B-col / C-col index
  const int c0      = halfSel * 8;        // A K-run base within a 32-K tile

  int tile = blockIdx.x * 8 + wid;
  const bool active = tile < NTILES;
  if (!active) tile = NTILES - 1;         // keep barriers convergent in tail
  _Float16* h = hbuf[wid];

  // Prebuilt B fragment: one aligned 32B load per lane (2x global_load_b128)
  auto loadB = [&](int f) {
    return *(const v16h*)(frags + ((size_t)(f * 32 + lane)) * 16);
  };
  // A fragment from a row-major f32 [N,32] buffer: 4x float4 + pk-cvt
  auto loadA_g = [&](const float* __restrict__ buf) {
    const float* r = buf + (size_t)(tile * 16 + mn) * DD + c0;
    const float4 q0 = *(const float4*)(r);
    const float4 q1 = *(const float4*)(r + 4);
    const float4 q2 = *(const float4*)(r + 16);
    const float4 q3 = *(const float4*)(r + 20);
    v16h a;
    a[0]  = (_Float16)q0.x; a[1]  = (_Float16)q0.y;
    a[2]  = (_Float16)q0.z; a[3]  = (_Float16)q0.w;
    a[4]  = (_Float16)q1.x; a[5]  = (_Float16)q1.y;
    a[6]  = (_Float16)q1.z; a[7]  = (_Float16)q1.w;
    a[8]  = (_Float16)q2.x; a[9]  = (_Float16)q2.y;
    a[10] = (_Float16)q2.z; a[11] = (_Float16)q2.w;
    a[12] = (_Float16)q3.x; a[13] = (_Float16)q3.y;
    a[14] = (_Float16)q3.z; a[15] = (_Float16)q3.w;
    return a;
  };
  // A fragment from the LDS f16 16x32 tile: 2x ds_load_b128
  auto loadA_l = [&]() {
    const _Float16* r = h + mn * HH + c0;
    const v8h lo = *(const v8h*)(r);
    const v8h hi = *(const v8h*)(r + 16);
    v16h a;
#pragma unroll
    for (int i = 0; i < 8; ++i) { a[i] = lo[i]; a[8 + i] = hi[i]; }
    return a;
  };
  // bias + tanh on the C fragment, write into LDS activation tile
  auto storeH = [&](v8f c, const float* __restrict__ bias, int n0) {
    const float bb = bias[n0 + mn];
#pragma unroll
    for (int v = 0; v < 8; ++v)
      h[(v + 8 * halfSel) * HH + n0 + mn] = (_Float16)fast_tanh(c[v] + bb);
  };

  // ---- layer 1: h = tanh([mi|mo|x] @ W1 + b1), K = 96 (3 K-tiles) ----
  {
    v16h a0 = loadA_g(mi), a1 = loadA_g(mo), a2 = loadA_g(x);
    v8f acc0 = {}, acc1 = {};
    acc0 = wmma_f16(a0, loadB(0), acc0);
    acc0 = wmma_f16(a1, loadB(1), acc0);
    acc0 = wmma_f16(a2, loadB(2), acc0);
    acc1 = wmma_f16(a0, loadB(3), acc1);
    acc1 = wmma_f16(a1, loadB(4), acc1);
    acc1 = wmma_f16(a2, loadB(5), acc1);
    storeH(acc0, b1, 0);
    storeH(acc1, b1, 16);
  }

  // ---- layers 2 & 3: h = tanh(h @ W + b), K = 32 ----
  const float* Bs[2] = { b2, b3 };
#pragma unroll
  for (int L = 0; L < 2; ++L) {
    __syncthreads();                 // RAW: LDS writes visible
    v16h a = loadA_l();
    __syncthreads();                 // WAR: reads done before overwrite
    v8f acc0 = {}, acc1 = {};
    acc0 = wmma_f16(a, loadB(6 + 2 * L), acc0);
    acc1 = wmma_f16(a, loadB(7 + 2 * L), acc1);
    storeH(acc0, Bs[L], 0);
    storeH(acc1, Bs[L], 16);
  }

  // ---- layer 4: out = tanh(h @ W4 + b4), written f32 to global ----
  __syncthreads();
  {
    v16h a = loadA_l();
    v8f acc0 = {}, acc1 = {};
    acc0 = wmma_f16(a, loadB(10), acc0);
    acc1 = wmma_f16(a, loadB(11), acc1);
    if (active) {
      const float bb0 = b4[mn], bb1 = b4[16 + mn];
#pragma unroll
      for (int v = 0; v < 8; ++v) {
        const int row = tile * 16 + v + 8 * halfSel;
        out[(size_t)row * HH + mn]      = fast_tanh(acc0[v] + bb0);
        out[(size_t)row * HH + 16 + mn] = fast_tanh(acc1[v] + bb1);
      }
    }
  }
}

// ---------------------------------------------------------------------------
// launch
// ---------------------------------------------------------------------------
extern "C" void kernel_launch(void* const* d_in, const int* in_sizes, int n_in,
                              void* d_out, int out_size, void* d_ws, size_t ws_size,
                              hipStream_t stream) {
  (void)in_sizes; (void)n_in; (void)out_size; (void)ws_size;

  const float* x   = (const float*)d_in[0];     // [N,32]
  const float* e   = (const float*)d_in[1];     // [E]
  const int*   ei  = (const int*)d_in[2];       // [2,E]
  const int*   src = ei;
  const int*   dst = ei + EE;
  const float* W1  = (const float*)d_in[3];     // [96,32]
  const float* B1  = (const float*)d_in[4];
  const float* W2  = (const float*)d_in[5];     // [32,32]
  const float* B2  = (const float*)d_in[6];
  const float* W3  = (const float*)d_in[7];
  const float* B3  = (const float*)d_in[8];
  const float* W4  = (const float*)d_in[9];
  const float* B4  = (const float*)d_in[10];
  float* out = (float*)d_out;                   // [N,32] f32

  // workspace: [mi f32 N*32][mo f32 N*32][B-fragments f16, 12*32*16]
  float*    mi    = (float*)d_ws;
  float*    mo    = mi + (size_t)NN * DD;
  _Float16* frags = (_Float16*)(mo + (size_t)NN * DD);   // 32B-aligned

  const int n4 = (2 * NN * DD) / 4;             // 800K float4s
  zero_kernel<<<(n4 + 255) / 256, 256, 0, stream>>>((float4*)mi, n4);
  build_frags_kernel<<<1, 384, 0, stream>>>(W1, W2, W3, W4, frags);
  edge_scatter_kernel<<<4096, 256, 0, stream>>>(x, e, src, dst, mi, mo);
  mlp_wmma_kernel<<<(NTILES + 7) / 8, 256, 0, stream>>>(
      mi, mo, x, frags, B1, B2, B3, B4, out);
}
